// DDiTBlock_65292092834414
// MI455X (gfx1250) — matrix operand order repaired and checked by reading, not compiled, not run on hardware
//
#include <hip/hip_runtime.h>
#include <math.h>
#include <stdint.h>

typedef __bf16 bf16;
typedef __attribute__((ext_vector_type(16))) __bf16 v16bf;
typedef __attribute__((ext_vector_type(8)))  float  v8f;
typedef int v4i __attribute__((__vector_size__(4 * sizeof(int))));

#define NB 4
#define SS 2048
#define DD 1024
#define HH 16
#define HD 64
#define DFF 4096
#define SIXD 6144
#define MROWS 8192   // B*S

#define ASG __attribute__((address_space(1)))
#define ASL __attribute__((address_space(3)))

union Frag { uint4 u[2]; v16bf v; };

// ---- CDNA5 async memory->LDS copy (ASYNCcnt-tracked), 16 bytes per lane ----
__device__ __forceinline__ void async_cp16(const bf16* g, bf16* l) {
#if __has_builtin(__builtin_amdgcn_global_load_async_to_lds_b128)
  __builtin_amdgcn_global_load_async_to_lds_b128(
      (ASG v4i*)(uintptr_t)g, (ASL v4i*)(uint32_t)(uintptr_t)l, 0, 0);
#else
  asm volatile("global_load_async_to_lds_b128 %0, %1, off"
               :: "v"((uint32_t)(uintptr_t)l), "v"(g) : "memory");
#endif
}
__device__ __forceinline__ void wait_async0() {
#if __has_builtin(__builtin_amdgcn_s_wait_asynccnt)
  __builtin_amdgcn_s_wait_asynccnt(0);
#else
  asm volatile("s_wait_asynccnt 0x0" ::: "memory");
#endif
}

// ---------------- small elementwise / setup kernels ----------------

// f32 [K][N] -> bf16 [N][K] (32x32 LDS-tiled transpose)
__global__ __launch_bounds__(256) void transpose_bf16_k(const float* __restrict__ in,
                                                        bf16* __restrict__ out,
                                                        int K, int N) {
  __shared__ float t[32][33];
  int n0 = blockIdx.x * 32, k0 = blockIdx.y * 32;
  int tx = threadIdx.x & 31, ty = threadIdx.x >> 5;   // ty 0..7
#pragma unroll
  for (int i = 0; i < 32; i += 8)
    t[ty + i][tx] = in[(size_t)(k0 + ty + i) * N + n0 + tx];
  __syncthreads();
#pragma unroll
  for (int i = 0; i < 32; i += 8)
    out[(size_t)(n0 + ty + i) * K + k0 + tx] = (bf16)t[tx][ty + i];
}

// mod[b][j] = c[b]@ada_W[:,j] + ada_b[j]
__global__ __launch_bounds__(256) void mod_kernel(const float* __restrict__ c,
                                                  const float* __restrict__ W,
                                                  const float* __restrict__ bia,
                                                  float* __restrict__ mod) {
  int j = blockIdx.x * 256 + threadIdx.x;
  int b = blockIdx.y;
  float acc = bia[j];
  const float* cb = c + b * 1024;
  for (int k = 0; k < 1024; ++k) acc += cb[k] * W[(size_t)k * SIXD + j];
  mod[b * SIXD + j] = acc;
}

// y = LN(x)*w*(1+sc)+sh  -> bf16
__global__ __launch_bounds__(256) void ln_mod_kernel(const float* __restrict__ x,
                                                     const float* __restrict__ w,
                                                     const float* __restrict__ mod,
                                                     int shOff, int scOff,
                                                     bf16* __restrict__ out) {
  __shared__ float r1[256], r2[256];
  int row = blockIdx.x;
  int b = row >> 11;
  int tid = threadIdx.x;
  const float* xr = x + (size_t)row * DD;
  float v[4], s = 0.f, s2 = 0.f;
#pragma unroll
  for (int i = 0; i < 4; ++i) { v[i] = xr[tid + i * 256]; s += v[i]; s2 += v[i] * v[i]; }
  r1[tid] = s; r2[tid] = s2; __syncthreads();
  for (int off = 128; off > 0; off >>= 1) {
    if (tid < off) { r1[tid] += r1[tid + off]; r2[tid] += r2[tid + off]; }
    __syncthreads();
  }
  float mu = r1[0] * (1.f / 1024.f);
  float var = r2[0] * (1.f / 1024.f) - mu * mu;
  float inv = rsqrtf(var + 1e-5f);
  const float* mrow = mod + b * SIXD;
#pragma unroll
  for (int i = 0; i < 4; ++i) {
    int col = tid + i * 256;
    float y = (v[i] - mu) * inv * w[col];
    y = y * (1.f + mrow[scOff + col]) + mrow[shOff + col];
    out[(size_t)row * DD + col] = (bf16)y;
  }
}

// RMSNorm + RoPE per head; one wave per (b,s,h); lane holds d and d+32.
// V is written TRANSPOSED: vo[((b*H+h)*HD+d)*S + s]
__global__ __launch_bounds__(256) void rope_kernel(const bf16* __restrict__ qkv,
                                                   const float* __restrict__ cosb,
                                                   const float* __restrict__ sinb,
                                                   const float* __restrict__ qw,
                                                   const float* __restrict__ kw,
                                                   bf16* __restrict__ qo,
                                                   bf16* __restrict__ ko,
                                                   bf16* __restrict__ vo) {
  int warp = blockIdx.x * 8 + (threadIdx.x >> 5);
  int lane = threadIdx.x & 31;
  int h = warp & 15;
  int bs = warp >> 4;              // b*S+s
  int s = bs & (SS - 1);
  int b = bs >> 11;
  const bf16* qp = qkv + (size_t)bs * 3072 + h * 64;
  const bf16* kp = qp + 1024;
  const bf16* vp = qp + 2048;
  int d1 = lane, d2 = lane + 32;
  float q1 = (float)qp[d1], q2 = (float)qp[d2];
  float k1 = (float)kp[d1], k2 = (float)kp[d2];
  float sq = q1 * q1 + q2 * q2, sk = k1 * k1 + k2 * k2;
#pragma unroll
  for (int m = 16; m >= 1; m >>= 1) { sq += __shfl_xor(sq, m, 32); sk += __shfl_xor(sk, m, 32); }
  float rq = rsqrtf(sq * (1.f / 64.f) + 1e-6f);
  float rk = rsqrtf(sk * (1.f / 64.f) + 1e-6f);
  q1 *= rq * qw[d1]; q2 *= rq * qw[d2];
  k1 *= rk * kw[d1]; k2 *= rk * kw[d2];
  float c1 = cosb[(size_t)s * HD + d1], c2 = cosb[(size_t)s * HD + d2];
  float s1 = sinb[(size_t)s * HD + d1], s2v = sinb[(size_t)s * HD + d2];
  float qo1 = q1 * c1 - q2 * s1, qo2 = q2 * c2 + q1 * s2v;
  float ko1 = k1 * c1 - k2 * s1, ko2 = k2 * c2 + k1 * s2v;
  size_t ob = (size_t)bs * DD + h * 64;
  qo[ob + d1] = (bf16)qo1; qo[ob + d2] = (bf16)qo2;
  ko[ob + d1] = (bf16)ko1; ko[ob + d2] = (bf16)ko2;
  size_t vtb = ((size_t)(b * HH + h) * HD) * SS + s;
  vo[vtb + (size_t)d1 * SS] = vp[d1];
  vo[vtb + (size_t)d2 * SS] = vp[d2];
}

// -------- WMMA GEMM: C[M,N] = A[M,K] @ Bt[N,K]^T (+epilogue) --------
// 128x128 block tile, 8 waves x (32x64), async LDS staging.
// MODE 0: out bf16 raw
// MODE 1: outF = xres + mod[g]*acc
// MODE 2: out bf16 = gelu(acc + bias)
// MODE 3: outF = xres + mod[g]*(acc + bias)
template <int MODE>
__global__ __launch_bounds__(256) void gemm_kernel(const bf16* __restrict__ A,
                                                   const bf16* __restrict__ Bt,
                                                   int N, int K,
                                                   float* __restrict__ outF,
                                                   bf16* __restrict__ outB,
                                                   const float* __restrict__ xres,
                                                   const float* __restrict__ mod,
                                                   int gOff,
                                                   const float* __restrict__ bias) {
  __shared__ bf16 As[128][40];   // [m][k], +8 pad keeps 16B align, kills conflicts
  __shared__ bf16 Bs[128][40];   // [n][k]
  int tid = threadIdx.x;
  int wave = tid >> 5, lane = tid & 31, hi = lane >> 4, lo = lane & 15;
  int m0 = blockIdx.y * 128, n0 = blockIdx.x * 128;
  int wm = (wave >> 1) * 32, wn = (wave & 1) * 64;
  int sr = tid >> 1, sc = (tid & 1) * 16;   // staging: 128 rows, 2 threads/row
  v8f acc[2][4];
#pragma unroll
  for (int i = 0; i < 2; ++i)
#pragma unroll
    for (int j = 0; j < 4; ++j) acc[i][j] = {};
  for (int k0 = 0; k0 < K; k0 += 32) {
    async_cp16(A  + (size_t)(m0 + sr) * K + k0 + sc,     &As[sr][sc]);
    async_cp16(A  + (size_t)(m0 + sr) * K + k0 + sc + 8, &As[sr][sc + 8]);
    async_cp16(Bt + (size_t)(n0 + sr) * K + k0 + sc,     &Bs[sr][sc]);
    async_cp16(Bt + (size_t)(n0 + sr) * K + k0 + sc + 8, &Bs[sr][sc + 8]);
    wait_async0();
    __syncthreads();
    Frag af[2];    // A 16x32: lane lo=row, k = (j>>3)*16 + hi*8 + (j&7)
#pragma unroll
    for (int i = 0; i < 2; ++i) {
      int row = wm + i * 16 + lo;
      af[i].u[0] = *(const uint4*)&As[row][hi * 8];
      af[i].u[1] = *(const uint4*)&As[row][16 + hi * 8];
    }
    Frag bfr[4];   // B 32x16: lane lo=col, k = (j>>3)*16 + hi*8 + (j&7)
#pragma unroll
    for (int j = 0; j < 4; ++j) {
      int col = wn + j * 16 + lo;
      bfr[j].u[0] = *(const uint4*)&Bs[col][hi * 8];
      bfr[j].u[1] = *(const uint4*)&Bs[col][16 + hi * 8];
    }
#pragma unroll
    for (int i = 0; i < 2; ++i)
#pragma unroll
      for (int j = 0; j < 4; ++j)
        acc[i][j] = __builtin_amdgcn_wmma_f32_16x16x32_bf16(
            false, af[i].v, false, bfr[j].v, (short)0, acc[i][j], false, false);
    __syncthreads();
  }
#pragma unroll
  for (int i = 0; i < 2; ++i)
#pragma unroll
    for (int j = 0; j < 4; ++j)
#pragma unroll
      for (int r = 0; r < 8; ++r) {
        int gm = m0 + wm + i * 16 + hi * 8 + r;
        int gn = n0 + wn + j * 16 + lo;
        size_t idx = (size_t)gm * N + gn;
        float a = acc[i][j][r];
        if constexpr (MODE == 0) {
          outB[idx] = (bf16)a;
        } else if constexpr (MODE == 1) {
          int b = gm >> 11;
          outF[idx] = xres[idx] + mod[b * SIXD + gOff + gn] * a;
        } else if constexpr (MODE == 2) {
          float v = a + bias[gn];
          v = 0.5f * v * (1.f + tanhf(0.7978845608f * (v + 0.044715f * v * v * v)));
          outB[idx] = (bf16)v;
        } else {
          int b = gm >> 11;
          outF[idx] = xres[idx] + mod[b * SIXD + gOff + gn] * (a + bias[gn]);
        }
      }
}

// ---- flash attention: WMMA QK^T + online softmax + WMMA P.V ----
// Q,K layout [B,S,H,HD]; V pre-transposed [B,H,HD,S]; out [B,S,H,HD]
__global__ __launch_bounds__(256) void attn_kernel(const bf16* __restrict__ Q,
                                                   const bf16* __restrict__ Kg,
                                                   const bf16* __restrict__ Vt_g,
                                                   bf16* __restrict__ Og) {
  __shared__ bf16 Ks[32][64];     // [key][d]
  __shared__ bf16 Vt[64][32];     // [d][key]
  __shared__ bf16 Ps[8][16][32];  // per-wave P tile [m][n]
  int bh = blockIdx.y;
  int b = bh >> 4, h = bh & 15;
  int tid = threadIdx.x, wave = tid >> 5, lane = tid & 31, hi = lane >> 4, lo = lane & 15;
  int q0 = blockIdx.x * 128 + wave * 16;
  const float scale = 0.125f;     // 1/sqrt(64)
  Frag qa[2];
  {
    size_t qb = ((size_t)(b * SS + q0 + lo) * HH + h) * HD;
#pragma unroll
    for (int c = 0; c < 2; ++c) {
      qa[c].u[0] = *(const uint4*)(Q + qb + c * 32 + hi * 8);
      qa[c].u[1] = *(const uint4*)(Q + qb + c * 32 + 16 + hi * 8);
    }
  }
  const bf16* vgbase = Vt_g + ((size_t)(b * HH + h) * HD) * SS;
  int kst_r = tid >> 3, kst_c = (tid & 7) * 8;   // K staging: 32 rows x 8 thr
  int vst_d = tid >> 2, vst_c = (tid & 3) * 8;   // V staging: 64 rows x 4 thr
  v8f o[4] = {};
  float rowM[8], rowL[8];
#pragma unroll
  for (int r = 0; r < 8; ++r) { rowM[r] = -3.0e38f; rowL[r] = 0.f; }

  for (int kc = 0; kc < SS; kc += 32) {
    async_cp16(Kg + ((size_t)(b * SS + kc + kst_r) * HH + h) * HD + kst_c, &Ks[kst_r][kst_c]);
    async_cp16(vgbase + (size_t)vst_d * SS + kc + vst_c, &Vt[vst_d][vst_c]);
    wait_async0();
    __syncthreads();
    v8f l0 = {}, l1 = {};
#pragma unroll
    for (int c = 0; c < 2; ++c) {
      Frag kb0, kb1;
      kb0.u[0] = *(const uint4*)&Ks[lo][c * 32 + hi * 8];
      kb0.u[1] = *(const uint4*)&Ks[lo][c * 32 + 16 + hi * 8];
      kb1.u[0] = *(const uint4*)&Ks[16 + lo][c * 32 + hi * 8];
      kb1.u[1] = *(const uint4*)&Ks[16 + lo][c * 32 + 16 + hi * 8];
      l0 = __builtin_amdgcn_wmma_f32_16x16x32_bf16(false, qa[c].v, false, kb0.v, (short)0, l0, false, false);
      l1 = __builtin_amdgcn_wmma_f32_16x16x32_bf16(false, qa[c].v, false, kb1.v, (short)0, l1, false, false);
    }
    float corr[8];
#pragma unroll
    for (int r = 0; r < 8; ++r) {
      float a0 = l0[r] * scale, a1 = l1[r] * scale;
      float mx = fmaxf(a0, a1);
      mx = fmaxf(mx, __shfl_xor(mx, 1, 32));
      mx = fmaxf(mx, __shfl_xor(mx, 2, 32));
      mx = fmaxf(mx, __shfl_xor(mx, 4, 32));
      mx = fmaxf(mx, __shfl_xor(mx, 8, 32));
      float mn = fmaxf(rowM[r], mx);
      float cf = __expf(rowM[r] - mn);
      float p0 = __expf(a0 - mn), p1 = __expf(a1 - mn);
      float sm = p0 + p1;
      sm += __shfl_xor(sm, 1, 32);
      sm += __shfl_xor(sm, 2, 32);
      sm += __shfl_xor(sm, 4, 32);
      sm += __shfl_xor(sm, 8, 32);
      rowL[r] = rowL[r] * cf + sm;
      rowM[r] = mn; corr[r] = cf;
      Ps[wave][hi * 8 + r][lo] = (bf16)p0;
      Ps[wave][hi * 8 + r][16 + lo] = (bf16)p1;
    }
#pragma unroll
    for (int t = 0; t < 4; ++t)
#pragma unroll
      for (int r = 0; r < 8; ++r) o[t][r] *= corr[r];
    __syncthreads();
    Frag pa;
    pa.u[0] = *(const uint4*)&Ps[wave][lo][hi * 8];
    pa.u[1] = *(const uint4*)&Ps[wave][lo][16 + hi * 8];
#pragma unroll
    for (int t = 0; t < 4; ++t) {
      Frag vb;
      vb.u[0] = *(const uint4*)&Vt[t * 16 + lo][hi * 8];
      vb.u[1] = *(const uint4*)&Vt[t * 16 + lo][16 + hi * 8];
      o[t] = __builtin_amdgcn_wmma_f32_16x16x32_bf16(false, pa.v, false, vb.v, (short)0, o[t], false, false);
    }
    __syncthreads();
  }
#pragma unroll
  for (int t = 0; t < 4; ++t)
#pragma unroll
    for (int r = 0; r < 8; ++r) {
      int m = hi * 8 + r;
      size_t dst = ((size_t)(b * SS + q0 + m) * HH + h) * HD + t * 16 + lo;
      Og[dst] = (bf16)(o[t][r] / rowL[r]);
    }
}

// ---------------- launch ----------------

#define OFF_WQKV  ((size_t)0)
#define OFF_WOUT  ((size_t)6291456)
#define OFF_W1    ((size_t)8388608)
#define OFF_W2    ((size_t)16777216)
#define OFF_MOD   ((size_t)25165824)
#define OFF_A     ((size_t)25264128)
#define OFF_BIG   ((size_t)42041344)
#define OFF_Q     ((size_t)109150208)
#define OFF_K     ((size_t)125927424)
#define OFF_V     ((size_t)142704640)
#define OFF_ATTN  ((size_t)159481856)

extern "C" void kernel_launch(void* const* d_in, const int* in_sizes, int n_in,
                              void* d_out, int out_size, void* d_ws, size_t ws_size,
                              hipStream_t stream) {
  const float* x    = (const float*)d_in[0];
  const float* cosb = (const float*)d_in[1];
  const float* sinb = (const float*)d_in[2];
  const float* c    = (const float*)d_in[3];
  const float* n1w  = (const float*)d_in[4];
  const float* Wqkv = (const float*)d_in[5];
  const float* Wout = (const float*)d_in[6];
  const float* qnw  = (const float*)d_in[7];
  const float* knw  = (const float*)d_in[8];
  const float* n2w  = (const float*)d_in[9];
  const float* W1   = (const float*)d_in[10];
  const float* b1   = (const float*)d_in[11];
  const float* W2   = (const float*)d_in[12];
  const float* b2   = (const float*)d_in[13];
  const float* adaW = (const float*)d_in[14];
  const float* adab = (const float*)d_in[15];
  float* out = (float*)d_out;
  char* ws = (char*)d_ws;
  bf16* wqkvT = (bf16*)(ws + OFF_WQKV);   // [3072][1024]
  bf16* woutT = (bf16*)(ws + OFF_WOUT);   // [1024][1024]
  bf16* w1T   = (bf16*)(ws + OFF_W1);     // [4096][1024]
  bf16* w2T   = (bf16*)(ws + OFF_W2);     // [1024][4096]
  float* modp = (float*)(ws + OFF_MOD);
  bf16* bufA  = (bf16*)(ws + OFF_A);
  bf16* bufBig= (bf16*)(ws + OFF_BIG);
  bf16* bufQ  = (bf16*)(ws + OFF_Q);
  bf16* bufK  = (bf16*)(ws + OFF_K);
  bf16* bufV  = (bf16*)(ws + OFF_V);      // transposed [B,H,HD,S]
  bf16* bufAt = (bf16*)(ws + OFF_ATTN);

  // weight transpose+convert: f32 [K][N] -> bf16 [N][K]
  transpose_bf16_k<<<dim3(3072 / 32, 1024 / 32), 256, 0, stream>>>(Wqkv, wqkvT, 1024, 3072);
  transpose_bf16_k<<<dim3(1024 / 32, 1024 / 32), 256, 0, stream>>>(Wout, woutT, 1024, 1024);
  transpose_bf16_k<<<dim3(4096 / 32, 1024 / 32), 256, 0, stream>>>(W1, w1T, 1024, 4096);
  transpose_bf16_k<<<dim3(1024 / 32, 4096 / 32), 256, 0, stream>>>(W2, w2T, 4096, 1024);
  // adaLN modulation
  mod_kernel<<<dim3(SIXD / 256, NB), 256, 0, stream>>>(c, adaW, adab, modp);
  // LN1 * (1+sc_msa) + sh_msa
  ln_mod_kernel<<<MROWS, 256, 0, stream>>>(x, n1w, modp, 0, 1024, bufA);
  // QKV GEMM -> bf16 [B,S,3,H,HD]
  gemm_kernel<0><<<dim3(3072 / 128, MROWS / 128), 256, 0, stream>>>(
      bufA, wqkvT, 3072, DD, nullptr, bufBig, nullptr, nullptr, 0, nullptr);
  // RMSNorm + RoPE + V split (V transposed)
  rope_kernel<<<(NB * SS * HH) / 8, 256, 0, stream>>>(bufBig, cosb, sinb, qnw, knw, bufQ, bufK, bufV);
  // attention
  attn_kernel<<<dim3(SS / 128, NB * HH), 256, 0, stream>>>(bufQ, bufK, bufV, bufAt);
  // out-proj + gated residual -> d_out (f32 x_mid)
  gemm_kernel<1><<<dim3(DD / 128, MROWS / 128), 256, 0, stream>>>(
      bufAt, woutT, DD, DD, out, nullptr, x, modp, 2048, nullptr);
  // LN2 * (1+sc_mlp) + sh_mlp
  ln_mod_kernel<<<MROWS, 256, 0, stream>>>(out, n2w, modp, 3072, 4096, bufA);
  // MLP up + GELU
  gemm_kernel<2><<<dim3(DFF / 128, MROWS / 128), 256, 0, stream>>>(
      bufA, w1T, DFF, DD, nullptr, bufBig, nullptr, nullptr, 0, b1);
  // MLP down + gated residual -> d_out final
  gemm_kernel<3><<<dim3(DD / 128, MROWS / 128), 256, 0, stream>>>(
      bufBig, w2T, DD, DFF, out, nullptr, out, modp, 5120, b2);
}
